// MIM_67997922230507
// MI455X (gfx1250) — matrix operand order repaired and checked by reading, not compiled, not run on hardware
//
#include <hip/hip_runtime.h>
#include <hip/hip_bf16.h>

typedef __attribute__((ext_vector_type(16))) __bf16 v16bf;
typedef __attribute__((ext_vector_type(8)))  float  v8f;

#define NEG_SLOPE 0.2f

// ---------- workspace layout (float offsets) ----------
constexpr size_t OFF_HP     = 0;          // 524288  hp  (B,L,H,DK) flat
constexpr size_t OFF_DP     = 524288;     // 131072  dp
constexpr size_t OFF_WHH    = 655360;     // 524288  whh
constexpr size_t OFF_SI     = 1179648;    // 16384   s_i
constexpr size_t OFF_SJ     = 1196032;    // 16384   s_j
constexpr size_t OFF_AGG    = 1212416;    // 524288  agg
constexpr size_t OFF_PRE    = 1736704;    // 524288  hp + lrelu(agg)
constexpr size_t OFF_HP2    = 2260992;    // 524288  hp2
constexpr size_t OFF_DP2    = 2785280;    // 131072  dp2
constexpr size_t OFF_TI     = 2916352;    // 262144  t_i (B,L,128)
constexpr size_t OFF_TJ     = 3178496;    // 262144  t_j
constexpr size_t OFF_TI2    = 3440640;    // 262144  Ti2 = t_i @ oute_w^T
constexpr size_t OFF_TJ2    = 3702784;    // 262144  Tj2
constexpr size_t OFF_WCOMB  = 3964928;    // 16384   Wcomb (128x128)
constexpr size_t OFF_CCOMB  = 3981312;    // 128
constexpr size_t OFF_BIAS2  = 3981440;    // 128
constexpr size_t OFF_CSE    = 3981568;    // 16
constexpr size_t OFF_WSEBF  = 3981584;    // 1024 floats = 2048 bf16 (ws_e 128x16)
constexpr size_t OFF_WSEPK  = 3982608;    // 1024 floats (packed B frags, 4 kt)
constexpr size_t OFF_WBIGBF = 3983632;    // 8192 floats = 16384 bf16 (Wbig 128x128)
constexpr size_t OFF_WBIGPK = 3991824;    // 8192 floats (packed B frags, 4 kt x 8 nt)

__device__ __forceinline__ unsigned short f2bf(float x) {
  unsigned int u = __float_as_uint(x);
  return (unsigned short)((u + 0x7FFFu + ((u >> 16) & 1u)) >> 16);
}

union Frag16 { v16bf v; unsigned short s[16]; uint4 q[2]; };

__device__ __forceinline__ v8f wmma_bf16(v16bf a, v16bf b, v8f c) {
  return __builtin_amdgcn_wmma_f32_16x16x32_bf16(false, a, false, b, (short)0, c,
                                                 false, false);
}

// Build a 16x32 bf16 A-fragment row slice for this lane using native bf16 casts.
// Lanes 0-15: halfs = K kb..kb+7 and kb+16..kb+23 (kb already includes hi*8).
__device__ __forceinline__ v16bf make_afrag(const float* __restrict__ erow, int kb) {
  float4 f0 = *(const float4*)(erow + kb);
  float4 f1 = *(const float4*)(erow + kb + 4);
  float4 f2 = *(const float4*)(erow + kb + 16);
  float4 f3 = *(const float4*)(erow + kb + 20);
  v16bf a;
  a[0]  = (__bf16)f0.x; a[1]  = (__bf16)f0.y; a[2]  = (__bf16)f0.z; a[3]  = (__bf16)f0.w;
  a[4]  = (__bf16)f1.x; a[5]  = (__bf16)f1.y; a[6]  = (__bf16)f1.z; a[7]  = (__bf16)f1.w;
  a[8]  = (__bf16)f2.x; a[9]  = (__bf16)f2.y; a[10] = (__bf16)f2.z; a[11] = (__bf16)f2.w;
  a[12] = (__bf16)f3.x; a[13] = (__bf16)f3.y; a[14] = (__bf16)f3.z; a[15] = (__bf16)f3.w;
  return a;
}

// ---------- generic small linear: out[r,n] = bias[n] + sum_k in[r,k]*W[n,k] ----------
__global__ void lin_kernel(const float* __restrict__ in, const float* __restrict__ W,
                           const float* __restrict__ bias, float* __restrict__ out,
                           int R, int K, int N) {
  int idx = blockIdx.x * blockDim.x + threadIdx.x;
  if (idx >= R * N) return;
  int r = idx / N, n = idx - r * N;
  const float* a = in + (size_t)r * K;
  const float* w = W + (size_t)n * K;
  float acc = bias ? bias[n] : 0.0f;
  for (int k = 0; k < K; k += 4) {
    float4 av = *(const float4*)(a + k);
    float4 wv = *(const float4*)(w + k);
    acc += av.x * wv.x + av.y * wv.y + av.z * wv.z + av.w * wv.w;
  }
  out[idx] = acc;
}

// ---------- whh[r,h,d] = Wh_b[d] + sum_k hp[r,h,k]*Wh_w[d,k] ----------
__global__ void whh_kernel(const float* __restrict__ hp, const float* __restrict__ Wh_w,
                           const float* __restrict__ Wh_b, float* __restrict__ whh) {
  int idx = blockIdx.x * blockDim.x + threadIdx.x;
  if (idx >= 2048 * 256) return;
  int d = idx & 31;
  const float* a = hp + (idx & ~31);  // row+head base
  const float* w = Wh_w + d * 32;
  float acc = Wh_b[d];
  for (int k = 0; k < 32; ++k) acc += a[k] * w[k];
  whh[idx] = acc;
}

// ---------- s_i / s_j per (row, head) ----------
__global__ void sij_kernel(const float* __restrict__ hp, const float* __restrict__ dp,
                           const float* __restrict__ attn, float* __restrict__ si,
                           float* __restrict__ sj) {
  int idx = blockIdx.x * blockDim.x + threadIdx.x;
  if (idx >= 2048 * 8) return;
  int r = idx >> 3, h = idx & 7;
  const float* hh = hp + (size_t)r * 256 + h * 32;
  const float* dd = dp + (size_t)r * 64 + h * 8;
  float vi = 0.f, vj = 0.f;
  for (int d = 0; d < 32; ++d) { vi += hh[d] * attn[d]; vj += hh[d] * attn[32 + d]; }
  for (int t = 0; t < 8; ++t) { vi += dd[t] * attn[80 + t]; vj += dd[t] * attn[88 + t]; }
  si[idx] = vi; sj[idx] = vj;
}

// ---------- ws_e[k,h] = sum_d a_e[d]*We_w[h*16+d,k] (bf16, padded to 16 cols) ----------
__global__ void wse_kernel(const float* __restrict__ We_w, const float* __restrict__ We_b,
                           const float* __restrict__ attn, unsigned short* __restrict__ wse_bf,
                           float* __restrict__ cse) {
  int idx = blockIdx.x * blockDim.x + threadIdx.x;
  if (idx >= 128 * 16) return;
  int k = idx >> 4, h = idx & 15;
  float v = 0.f;
  if (h < 8) for (int d = 0; d < 16; ++d) v += attn[64 + d] * We_w[(h * 16 + d) * 128 + k];
  wse_bf[k * 16 + h] = f2bf(v);
  if (k == 0 && h < 8) {
    float c = 0.f;
    for (int d = 0; d < 16; ++d) c += attn[64 + d] * We_b[h * 16 + d];
    cse[h] = c;
  }
}

// ---------- Wcomb[c,k] = sum_d W_e[e,d]*We2_w[h*16+d,k], c=h*16+e ----------
__global__ void wcomb_kernel(const float* __restrict__ elw, const float* __restrict__ We2_w,
                             const float* __restrict__ We2_b, float* __restrict__ Wcomb,
                             float* __restrict__ ccomb) {
  int idx = blockIdx.x * blockDim.x + threadIdx.x;
  if (idx >= 128 * 128) return;
  int c = idx >> 7, k = idx & 127;
  int h = c >> 4, e = c & 15;
  float v = 0.f;
  for (int d = 0; d < 16; ++d) v += elw[e * 96 + d] * We2_w[(h * 16 + d) * 128 + k];
  Wcomb[c * 128 + k] = v;
  if (k == 0) {
    float cc = 0.f;
    for (int d = 0; d < 16; ++d) cc += elw[e * 96 + d] * We2_b[h * 16 + d];
    ccomb[c] = cc;
  }
}

// ---------- Wbig[k,f] = sum_c Wcomb[c,k]*oute_w[f,c];  bias2[f] ----------
__global__ void wbig_kernel(const float* __restrict__ Wcomb, const float* __restrict__ ccomb,
                            const float* __restrict__ elb, const float* __restrict__ oute_w,
                            const float* __restrict__ oute_b, unsigned short* __restrict__ wbig_bf,
                            float* __restrict__ bias2) {
  int idx = blockIdx.x * blockDim.x + threadIdx.x;
  if (idx >= 128 * 128) return;
  int k = idx >> 7, f = idx & 127;
  float v = 0.f;
  for (int c = 0; c < 128; ++c) v += Wcomb[c * 128 + k] * oute_w[f * 128 + c];
  wbig_bf[k * 128 + f] = f2bf(v);
  if (k == 0) {
    float bb = oute_b[f];
    for (int c = 0; c < 128; ++c) bb += (ccomb[c] + elb[c & 15]) * oute_w[f * 128 + c];
    bias2[f] = bb;
  }
}

// ---------- pack K x N bf16 matrix into per-lane WMMA B fragments ----------
// frag (kt,nt): lane<16 -> col=nt*16+lane, halfs t=0..7 = K kt*32+t, t=8..15 = K kt*32+16+(t-8)
//               lane>=16 -> col=nt*16+lane-16, halfs t=0..7 = K kt*32+8+t, t=8..15 = K kt*32+24+(t-8)
__global__ void packb_kernel(const unsigned short* __restrict__ src,
                             unsigned short* __restrict__ dst,
                             int Ktiles, int Ntiles, int Ncols) {
  int idx = blockIdx.x * blockDim.x + threadIdx.x;
  int total = Ktiles * Ntiles * 512;
  if (idx >= total) return;
  int t = idx & 15;
  int lane = (idx >> 4) & 31;
  int frag = idx >> 9;
  int kt = frag / Ntiles, nt = frag - kt * Ntiles;
  int k = kt * 32 + ((lane >= 16) ? 8 : 0) + (t & 7) + ((t >= 8) ? 16 : 0);
  int col = nt * 16 + (lane & 15);
  dst[idx] = src[k * Ncols + col];
}

// ---------- pass 1: s_e via WMMA -> lrelu/mask -> softmax -> agg ----------
__global__ void pass1_kernel(const float* __restrict__ edge,
                             const unsigned short* __restrict__ wse_pk,
                             const float* __restrict__ cse,
                             const float* __restrict__ si, const float* __restrict__ sj,
                             const float* __restrict__ whh, float* __restrict__ agg) {
  __shared__ float sc[256 * 8];
  int bi = blockIdx.x;                 // b*256 + i
  int b = bi >> 8, i = bi & 255;
  int tid = threadIdx.x;
  int wave = tid >> 5, lane = tid & 31;
  int rowLane = lane & 15, hi = lane >> 4;

  for (int tt = 0; tt < 2; ++tt) {
    int jt = wave + tt * 8;
    const float* erow = edge + ((size_t)bi * 256 + jt * 16 + rowLane) * 128;
    v8f acc = {0.f, 0.f, 0.f, 0.f, 0.f, 0.f, 0.f, 0.f};
#pragma unroll
    for (int kt = 0; kt < 4; ++kt) {
      v16bf a = make_afrag(erow, kt * 32 + hi * 8);
      Frag16 bf;
      const uint4* bp = (const uint4*)(wse_pk + (size_t)(kt * 32 + lane) * 16);
      bf.q[0] = bp[0]; bf.q[1] = bp[1];
      acc = wmma_bf16(a, bf.v, acc);
    }
    int h = rowLane;                   // C layout: col = lane&15
    if (h < 8) {
      float base = si[bi * 8 + h] + cse[h];
#pragma unroll
      for (int v = 0; v < 8; ++v) {
        int j = jt * 16 + v + hi * 8;
        float s = acc[v] + base + sj[(b * 256 + j) * 8 + h];
        s = (s >= 0.f) ? s : NEG_SLOPE * s;
        if (j == i) s = -3.4e38f;
        sc[j * 8 + h] = s;
      }
    }
  }
  __syncthreads();

  // wave == head; lane strided over j for reductions, then lane == dim d (DK==32)
  int h = wave;
  float m = -3.4e38f;
  for (int j = lane; j < 256; j += 32) m = fmaxf(m, sc[j * 8 + h]);
#pragma unroll
  for (int off = 16; off > 0; off >>= 1) m = fmaxf(m, __shfl_xor(m, off, 32));
  float Z = 0.f;
  for (int j = lane; j < 256; j += 32) Z += __expf(sc[j * 8 + h] - m);
#pragma unroll
  for (int off = 16; off > 0; off >>= 1) Z += __shfl_xor(Z, off, 32);
  float invZ = 1.0f / Z;
  // normalize in-place: sc now holds probabilities for this head's column
  for (int j = lane; j < 256; j += 32) sc[j * 8 + h] = __expf(sc[j * 8 + h] - m) * invZ;
  __syncthreads();
  float accd = 0.f;
  const float* wb = whh + (size_t)b * 65536 + h * 32 + lane;
  for (int j = 0; j < 256; ++j) {
    accd += sc[j * 8 + h] * wb[(size_t)j * 256];
  }
  agg[((size_t)bi * 8 + h) * 32 + lane] = accd;
}

// ---------- pre = hp + lrelu(agg) ----------
__global__ void pre_kernel(const float* __restrict__ hp, const float* __restrict__ agg,
                           float* __restrict__ pre, int n) {
  int idx = blockIdx.x * blockDim.x + threadIdx.x;
  if (idx >= n) return;
  float g = agg[idx];
  g = (g >= 0.f) ? g : NEG_SLOPE * g;
  pre[idx] = hp[idx] + g;
}

// ---------- t_i/t_j per (row, c=h*16+e) ----------
__global__ void titj_kernel(const float* __restrict__ hp2, const float* __restrict__ dp2,
                            const float* __restrict__ elw, float* __restrict__ ti,
                            float* __restrict__ tj) {
  int idx = blockIdx.x * blockDim.x + threadIdx.x;
  if (idx >= 2048 * 128) return;
  int r = idx >> 7, c = idx & 127;
  int h = c >> 4, e = c & 15;
  const float* hh = hp2 + (size_t)r * 256 + h * 32;
  const float* dd = dp2 + (size_t)r * 64 + h * 8;
  const float* wi = elw + e * 96 + 16;
  const float* wj = elw + e * 96 + 48;
  const float* wdi = elw + e * 96 + 80;
  const float* wdj = elw + e * 96 + 88;
  float vi = 0.f, vj = 0.f;
  for (int d = 0; d < 32; ++d) { vi += hh[d] * wi[d]; vj += hh[d] * wj[d]; }
  for (int t = 0; t < 8; ++t) { vi += dd[t] * wdi[t]; vj += dd[t] * wdj[t]; }
  ti[idx] = vi; tj[idx] = vj;
}

// ---------- pass 2: new_edge = edge @ Wbig + Ti2 + Tj2 + bias2 (diag -> oute_b) ----------
__global__ void pass2_kernel(const float* __restrict__ edge,
                             const unsigned short* __restrict__ wbig_pk,
                             const float* __restrict__ Ti2, const float* __restrict__ Tj2,
                             const float* __restrict__ bias2, const float* __restrict__ oute_b,
                             float* __restrict__ out_edge) {
  int tid = threadIdx.x;
  int wave = tid >> 5, lane = tid & 31;
  int rowLane = lane & 15, hi = lane >> 4;
  size_t row0 = ((size_t)blockIdx.x * 8 + wave) * 16;   // flattened (b,i,j)
  int b = (int)(row0 >> 16);
  int i = (int)((row0 >> 8) & 255);
  int j0 = (int)(row0 & 255);

  const float* erow = edge + (row0 + rowLane) * 128;
  v8f zero = {0.f, 0.f, 0.f, 0.f, 0.f, 0.f, 0.f, 0.f};
  v8f acc[8];
#pragma unroll
  for (int nt = 0; nt < 8; ++nt) acc[nt] = zero;

#pragma unroll
  for (int kt = 0; kt < 4; ++kt) {
    v16bf a = make_afrag(erow, kt * 32 + hi * 8);
#pragma unroll
    for (int nt = 0; nt < 8; ++nt) {
      Frag16 bf;
      const uint4* bp = (const uint4*)(wbig_pk + (size_t)((kt * 8 + nt) * 32 + lane) * 16);
      bf.q[0] = bp[0]; bf.q[1] = bp[1];
      acc[nt] = wmma_bf16(a, bf.v, acc[nt]);
    }
  }

  const float* tirow = Ti2 + ((size_t)b * 256 + i) * 128;
#pragma unroll
  for (int nt = 0; nt < 8; ++nt) {
    int col = nt * 16 + rowLane;
    float tv = tirow[col] + bias2[col];
    float ob = oute_b[col];
#pragma unroll
    for (int v = 0; v < 8; ++v) {
      int j = j0 + v + hi * 8;
      float val = acc[nt][v] + tv + Tj2[((size_t)b * 256 + j) * 128 + col];
      if (j == i) val = ob;
      out_edge[(row0 + v + hi * 8) * 128 + col] = val;
    }
  }
}

extern "C" void kernel_launch(void* const* d_in, const int* in_sizes, int n_in,
                              void* d_out, int out_size, void* d_ws, size_t ws_size,
                              hipStream_t stream) {
  (void)in_sizes; (void)n_in; (void)out_size; (void)ws_size;
  const float* node  = (const float*)d_in[0];
  const float* edge  = (const float*)d_in[1];
  const float* diff  = (const float*)d_in[2];
  const float* Wn_w  = (const float*)d_in[3];
  const float* Wn_b  = (const float*)d_in[4];
  const float* Wh_w  = (const float*)d_in[5];
  const float* Wh_b  = (const float*)d_in[6];
  const float* We_w  = (const float*)d_in[7];
  const float* We_b  = (const float*)d_in[8];
  const float* Wn2_w = (const float*)d_in[9];
  const float* Wn2_b = (const float*)d_in[10];
  const float* We2_w = (const float*)d_in[11];
  const float* We2_b = (const float*)d_in[12];
  const float* Wd_w  = (const float*)d_in[13];
  const float* Wd_b  = (const float*)d_in[14];
  const float* Wd2_w = (const float*)d_in[15];
  const float* Wd2_b = (const float*)d_in[16];
  const float* attn  = (const float*)d_in[17];
  const float* elw   = (const float*)d_in[18];
  const float* elb   = (const float*)d_in[19];
  const float* outn_w = (const float*)d_in[20];
  const float* outn_b = (const float*)d_in[21];
  const float* oute_w = (const float*)d_in[22];
  const float* oute_b = (const float*)d_in[23];

  float* ws = (float*)d_ws;
  float* hp    = ws + OFF_HP;
  float* dp    = ws + OFF_DP;
  float* whh   = ws + OFF_WHH;
  float* si    = ws + OFF_SI;
  float* sj    = ws + OFF_SJ;
  float* agg   = ws + OFF_AGG;
  float* pre   = ws + OFF_PRE;
  float* hp2   = ws + OFF_HP2;
  float* dp2   = ws + OFF_DP2;
  float* ti    = ws + OFF_TI;
  float* tj    = ws + OFF_TJ;
  float* Ti2   = ws + OFF_TI2;
  float* Tj2   = ws + OFF_TJ2;
  float* Wcomb = ws + OFF_WCOMB;
  float* ccomb = ws + OFF_CCOMB;
  float* bias2 = ws + OFF_BIAS2;
  float* cse   = ws + OFF_CSE;
  unsigned short* wse_bf  = (unsigned short*)(ws + OFF_WSEBF);
  unsigned short* wse_pk  = (unsigned short*)(ws + OFF_WSEPK);
  unsigned short* wbig_bf = (unsigned short*)(ws + OFF_WBIGBF);
  unsigned short* wbig_pk = (unsigned short*)(ws + OFF_WBIGPK);

  float* out_node = (float*)d_out;
  float* out_edge = (float*)d_out + 524288;

  // node-side precompute
  lin_kernel<<<2048, 256, 0, stream>>>(node, Wn_w, Wn_b, hp, 2048, 256, 256);
  lin_kernel<<<512, 256, 0, stream>>>(diff, Wd_w, Wd_b, dp, 2048, 64, 64);
  whh_kernel<<<2048, 256, 0, stream>>>(hp, Wh_w, Wh_b, whh);
  sij_kernel<<<64, 256, 0, stream>>>(hp, dp, attn, si, sj);

  // folded edge weights + WMMA fragment packing
  wse_kernel<<<8, 256, 0, stream>>>(We_w, We_b, attn, wse_bf, cse);
  packb_kernel<<<8, 256, 0, stream>>>(wse_bf, wse_pk, 4, 1, 16);
  wcomb_kernel<<<64, 256, 0, stream>>>(elw, We2_w, We2_b, Wcomb, ccomb);
  wbig_kernel<<<64, 256, 0, stream>>>(Wcomb, ccomb, elb, oute_w, oute_b, wbig_bf, bias2);
  packb_kernel<<<64, 256, 0, stream>>>(wbig_bf, wbig_pk, 4, 8, 128);

  // pass 1: attention over edges
  pass1_kernel<<<2048, 256, 0, stream>>>(edge, wse_pk, cse, si, sj, whh, agg);

  // node update
  pre_kernel<<<2048, 256, 0, stream>>>(hp, agg, pre, 524288);
  lin_kernel<<<2048, 256, 0, stream>>>(pre, outn_w, outn_b, out_node, 2048, 256, 256);
  lin_kernel<<<2048, 256, 0, stream>>>(out_node, Wn2_w, Wn2_b, hp2, 2048, 256, 256);
  lin_kernel<<<512, 256, 0, stream>>>(diff, Wd2_w, Wd2_b, dp2, 2048, 64, 64);
  titj_kernel<<<1024, 256, 0, stream>>>(hp2, dp2, elw, ti, tj);
  lin_kernel<<<1024, 256, 0, stream>>>(ti, oute_w, nullptr, Ti2, 2048, 128, 128);
  lin_kernel<<<1024, 256, 0, stream>>>(tj, oute_w, nullptr, Tj2, 2048, 128, 128);

  // pass 2: fused edge output GEMM
  pass2_kernel<<<4096, 256, 0, stream>>>(edge, wbig_pk, Ti2, Tj2, bias2, oute_b, out_edge);
}
